// GroupedQueryAttention_35141422416412
// MI455X (gfx1250) — compile-verified, matrix operand order checked by reading, not run on hardware
//
#include <hip/hip_runtime.h>
#include <hip/hip_bf16.h>

// ---------------------------------------------------------------------------
// GQA for MI455X (gfx1250, wave32, WMMA bf16 16x16x32, f32 accumulate)
//   D_MODEL=2048, NQ=32, NKV=8, HD=64, KVD=512, B=2, S=2048, M = B*S = 4096
// Pipeline:
//   1) convert activations f32->bf16
//   2) transpose+convert weights W[K,N] -> Wt[N,K] bf16
//   3) WMMA GEMM (async global->LDS double-buffered staging):
//        Q = q@Wq+bq (bf16), K = k@Wk+bk (bf16),
//        V = v@Wv+bv (bf16, stored transposed per (b,kv) head)
//   4) flash-attention per (b, head, 16-query tile), WMMA QK^T and PV
//   5) WMMA GEMM: out = attn@Wo+bo (f32 out)
// ---------------------------------------------------------------------------

#if defined(__has_builtin)
#  if __has_builtin(__builtin_amdgcn_global_load_async_to_lds_b128) && \
      __has_builtin(__builtin_amdgcn_s_wait_asynccnt)
#    define GEMM_ASYNC 1
#  endif
#endif
#ifndef GEMM_ASYNC
#  define GEMM_ASYNC 0
#endif

typedef __attribute__((ext_vector_type(16))) __bf16 v16bf;
typedef __attribute__((ext_vector_type(8)))  float  v8f;
typedef __attribute__((ext_vector_type(4)))  int    v4i;

union Frag { v16bf v; unsigned int u[8]; };

__device__ __forceinline__ unsigned short f2bf(float x) {
    unsigned int u = __builtin_bit_cast(unsigned int, x);
    u += 0x7FFFu + ((u >> 16) & 1u);      // round-to-nearest-even
    return (unsigned short)(u >> 16);
}

__device__ __forceinline__ v8f wmma_bf16(const v16bf& a, const v16bf& b, const v8f& c) {
    return __builtin_amdgcn_wmma_f32_16x16x32_bf16(
        false, a, false, b, (short)0, c, false, false);
}

__device__ __forceinline__ float rowmax16(float v) {
    v = fmaxf(v, __shfl_xor(v, 1, 32));
    v = fmaxf(v, __shfl_xor(v, 2, 32));
    v = fmaxf(v, __shfl_xor(v, 4, 32));
    v = fmaxf(v, __shfl_xor(v, 8, 32));
    return v;
}
__device__ __forceinline__ float rowsum16(float v) {
    v += __shfl_xor(v, 1, 32);
    v += __shfl_xor(v, 2, 32);
    v += __shfl_xor(v, 4, 32);
    v += __shfl_xor(v, 8, 32);
    return v;
}

// 16-byte global -> LDS copy: async (ASYNCcnt-tracked) on gfx1250, else sync.
__device__ __forceinline__ void copy16_g2l(const unsigned short* g, unsigned short* l) {
#if GEMM_ASYNC
    __builtin_amdgcn_global_load_async_to_lds_b128(
        (__attribute__((address_space(1))) v4i*)g,
        (__attribute__((address_space(3))) v4i*)l, 0, 0);
#else
    *(uint4*)l = *(const uint4*)g;
#endif
}
template <int N>
__device__ __forceinline__ void wait_async_le() {
#if GEMM_ASYNC
    __builtin_amdgcn_s_wait_asynccnt(N);
#endif
}

// ---------------------------------------------------------------------------
__global__ __launch_bounds__(256) void cvt_bf16_kernel(const float* __restrict__ in,
                                                       unsigned short* __restrict__ out,
                                                       int n) {
    int i = blockIdx.x * blockDim.x + threadIdx.x;
    if (i < n) out[i] = f2bf(in[i]);
}

// Wt[n*K + k] = bf16(W[k*N + n])
__global__ __launch_bounds__(256) void transpose_bf16_kernel(const float* __restrict__ W,
                                                             unsigned short* __restrict__ Wt,
                                                             int K, int N) {
    int i = blockIdx.x * blockDim.x + threadIdx.x;
    if (i < K * N) {
        int n = i / K;
        int k = i - n * K;
        Wt[i] = f2bf(W[(size_t)k * N + n]);
    }
}

// ---------------------------------------------------------------------------
// C[M,N] = A[M,K] (bf16) @ Wt[N,K]^T (bf16) + bias
// MODE 0: bf16 out, natural [M,N]
// MODE 1: f32  out, natural [M,N]
// MODE 2: bf16 out, V-transposed: out[((b*8+kv)*64+d)*2048 + s]
// block = 256 threads = 8 waves (4 x 2); block tile 128x64, wave tile 32x32.
// A/B k-tiles staged global->LDS with async b128 copies, double-buffered;
// LDS rows padded to 40 elements (80B) to spread ds_load_b128 across banks.
// ---------------------------------------------------------------------------
#define LDP 40   // padded LDS row stride in bf16 elements

template <int MODE>
__global__ __launch_bounds__(256) void gemm_bf16_kernel(
        const unsigned short* __restrict__ A,
        const unsigned short* __restrict__ Wt,
        const float* __restrict__ bias,
        void* __restrict__ outp,
        int M, int N, int K) {
    __shared__ __align__(16) unsigned short Alds[2][128 * LDP];
    __shared__ __align__(16) unsigned short Blds[2][64 * LDP];

    const int t     = threadIdx.x;
    const int lane  = t & 31;
    const int w     = t >> 5;
    const int wm    = w >> 1, wn = w & 1;
    const int blkM  = blockIdx.x * 128;
    const int blkN  = blockIdx.y * 64;
    const int tileM = blkM + wm * 32;
    const int tileN = blkN + wn * 32;
    const int mrow  = lane & 15;
    const int khalf = (lane >> 4) * 8;   // A/B fragment K split per ISA layout

    // staging chunk decomposition: 16B chunks, 4 per 32-element row
    const int ar0 = t >> 2,         ac0 = (t & 3) * 8;          // A chunk 0
    const int ar1 = (t + 256) >> 2, ac1 = ((t + 256) & 3) * 8;  // A chunk 1
    const int br  = t >> 2,         bc  = (t & 3) * 8;          // B chunk (rows 0..63)

    v8f c00 = {}, c01 = {}, c10 = {}, c11 = {};

    // prologue: stage k-tile 0 into buffer 0
    copy16_g2l(A  + (size_t)(blkM + ar0) * K + ac0, &Alds[0][ar0 * LDP + ac0]);
    copy16_g2l(A  + (size_t)(blkM + ar1) * K + ac1, &Alds[0][ar1 * LDP + ac1]);
    copy16_g2l(Wt + (size_t)(blkN + br ) * K + bc,  &Blds[0][br  * LDP + bc ]);

    int buf = 0;
    for (int kk = 0; kk < K; kk += 32) {
        const bool more = (kk + 32) < K;
        if (more) {
            const int nk = kk + 32;
            copy16_g2l(A  + (size_t)(blkM + ar0) * K + nk + ac0,
                       &Alds[buf ^ 1][ar0 * LDP + ac0]);
            copy16_g2l(A  + (size_t)(blkM + ar1) * K + nk + ac1,
                       &Alds[buf ^ 1][ar1 * LDP + ac1]);
            copy16_g2l(Wt + (size_t)(blkN + br ) * K + nk + bc,
                       &Blds[buf ^ 1][br  * LDP + bc ]);
            wait_async_le<3>();   // drain current tile (async completes in order)
        } else {
            wait_async_le<0>();
        }
        __syncthreads();

        const unsigned short* aL0 = &Alds[buf][(wm * 32      + mrow) * LDP + khalf];
        const unsigned short* aL1 = &Alds[buf][(wm * 32 + 16 + mrow) * LDP + khalf];
        const unsigned short* bL0 = &Blds[buf][(wn * 32      + mrow) * LDP + khalf];
        const unsigned short* bL1 = &Blds[buf][(wn * 32 + 16 + mrow) * LDP + khalf];
        Frag a0, a1, b0, b1;
        ((uint4*)a0.u)[0] = *(const uint4*)(aL0);
        ((uint4*)a0.u)[1] = *(const uint4*)(aL0 + 16);
        ((uint4*)a1.u)[0] = *(const uint4*)(aL1);
        ((uint4*)a1.u)[1] = *(const uint4*)(aL1 + 16);
        ((uint4*)b0.u)[0] = *(const uint4*)(bL0);
        ((uint4*)b0.u)[1] = *(const uint4*)(bL0 + 16);
        ((uint4*)b1.u)[0] = *(const uint4*)(bL1);
        ((uint4*)b1.u)[1] = *(const uint4*)(bL1 + 16);

        c00 = wmma_bf16(a0.v, b0.v, c00);
        c01 = wmma_bf16(a0.v, b1.v, c01);
        c10 = wmma_bf16(a1.v, b0.v, c10);
        c11 = wmma_bf16(a1.v, b1.v, c11);

        __syncthreads();   // protect buffer being read from next overwrite
        buf ^= 1;
    }

    const int rowOff = (lane >> 4) * 8;   // C layout: upper 16 lanes hold rows +8
    const int col    = lane & 15;
    const v8f* acc[4] = { &c00, &c01, &c10, &c11 };
    #pragma unroll
    for (int q = 0; q < 4; q++) {
        const int baseM = tileM + ((q >> 1) ? 16 : 0);
        const int baseN = tileN + ((q & 1) ? 16 : 0);
        const int cc    = baseN + col;
        const float bv  = bias[cc];
        #pragma unroll
        for (int r = 0; r < 8; r++) {
            const int row = baseM + rowOff + r;
            const float val = (*acc[q])[r] + bv;
            if (MODE == 0) {
                ((unsigned short*)outp)[(size_t)row * N + cc] = f2bf(val);
            } else if (MODE == 1) {
                ((float*)outp)[(size_t)row * N + cc] = val;
            } else {
                const int bb = row >> 11, ss = row & 2047;
                const int kvh = cc >> 6,  dd = cc & 63;
                ((unsigned short*)outp)[(((size_t)bb * 8 + kvh) * 64 + dd) * 2048 + ss]
                    = f2bf(val);
            }
        }
    }
}

// ---------------------------------------------------------------------------
// Flash attention: 1 wave per (b, qhead, 16-query tile); key tiles of 32.
//  Qp  : bf16 [B*S, 2048]            (row-major)
//  Kp  : bf16 [B*S, 512]             (row-major)
//  VpT : bf16 [B][8][64][2048]       (d-major, keys contiguous)
//  out : bf16 [B*S, 2048]
// ---------------------------------------------------------------------------
__global__ __launch_bounds__(32) void gqa_attn_kernel(
        const unsigned short* __restrict__ Qp,
        const unsigned short* __restrict__ Kp,
        const unsigned short* __restrict__ VpT,
        unsigned short* __restrict__ attnOut) {
    __shared__ __align__(16) unsigned short plds[16 * 32];   // P tile re-swizzle

    const int lane  = threadIdx.x;
    const int qt    = blockIdx.x & 127;
    const int h     = (blockIdx.x >> 7) & 31;
    const int b     = blockIdx.x >> 12;
    const int kv    = h & 7;                 // q head (g*8 + j) -> kv head j
    const int mrow  = lane & 15;
    const int khalf = (lane >> 4) * 8;
    const int rowOff = (lane >> 4) * 8;

    // Q A-fragments (persist): rows qt*16+mrow, d 0..31 and 32..63
    const unsigned short* qbase =
        Qp + ((size_t)(b * 2048 + qt * 16 + mrow)) * 2048 + h * 64;
    Frag qa0, qa1;
    ((uint4*)qa0.u)[0] = *(const uint4*)(qbase + khalf);
    ((uint4*)qa0.u)[1] = *(const uint4*)(qbase + khalf + 16);
    ((uint4*)qa1.u)[0] = *(const uint4*)(qbase + 32 + khalf);
    ((uint4*)qa1.u)[1] = *(const uint4*)(qbase + 32 + khalf + 16);

    v8f o0 = {}, o1 = {}, o2 = {}, o3 = {};
    float m[8], l[8];
    #pragma unroll
    for (int r = 0; r < 8; r++) { m[r] = -1e30f; l[r] = 0.0f; }

    const unsigned short* kbase = Kp  + ((size_t)b * 2048) * 512 + kv * 64;
    const unsigned short* vbase = VpT + ((size_t)(b * 8 + kv) * 64) * 2048;
    const float scale = 0.125f;              // 1/sqrt(64)

    for (int kt = 0; kt < 2048; kt += 32) {
        // K^T B-fragments: N = key, contraction = d (contiguous in Kp rows)
        const unsigned short* k0p = kbase + (size_t)(kt      + mrow) * 512;
        const unsigned short* k1p = kbase + (size_t)(kt + 16 + mrow) * 512;
        Frag kb00, kb01, kb10, kb11;
        ((uint4*)kb00.u)[0] = *(const uint4*)(k0p + khalf);
        ((uint4*)kb00.u)[1] = *(const uint4*)(k0p + khalf + 16);
        ((uint4*)kb01.u)[0] = *(const uint4*)(k0p + 32 + khalf);
        ((uint4*)kb01.u)[1] = *(const uint4*)(k0p + 32 + khalf + 16);
        ((uint4*)kb10.u)[0] = *(const uint4*)(k1p + khalf);
        ((uint4*)kb10.u)[1] = *(const uint4*)(k1p + khalf + 16);
        ((uint4*)kb11.u)[0] = *(const uint4*)(k1p + 32 + khalf);
        ((uint4*)kb11.u)[1] = *(const uint4*)(k1p + 32 + khalf + 16);

        v8f s0 = {}, s1 = {};
        s0 = wmma_bf16(qa0.v, kb00.v, s0);
        s0 = wmma_bf16(qa1.v, kb01.v, s0);
        s1 = wmma_bf16(qa0.v, kb10.v, s1);
        s1 = wmma_bf16(qa1.v, kb11.v, s1);

        // online softmax on C-layout rows (row spans 16 lanes of a half-wave)
        #pragma unroll
        for (int r = 0; r < 8; r++) {
            const float sv0 = s0[r] * scale;
            const float sv1 = s1[r] * scale;
            const float rmax = rowmax16(fmaxf(sv0, sv1));
            const float mn = fmaxf(m[r], rmax);
            const float alpha = __expf(m[r] - mn);
            m[r] = mn;
            const float e0 = __expf(sv0 - mn);
            const float e1 = __expf(sv1 - mn);
            l[r] = l[r] * alpha + rowsum16(e0 + e1);
            o0[r] *= alpha; o1[r] *= alpha; o2[r] *= alpha; o3[r] *= alpha;
            const int prow = rowOff + r;
            plds[prow * 32 + mrow]      = f2bf(e0);
            plds[prow * 32 + 16 + mrow] = f2bf(e1);
        }
        __syncthreads();   // C-layout -> A-layout re-swizzle through LDS

        Frag pa;
        ((uint4*)pa.u)[0] = *(const uint4*)(&plds[mrow * 32 + khalf]);
        ((uint4*)pa.u)[1] = *(const uint4*)(&plds[mrow * 32 + khalf + 16]);

        // V B-fragments: N = d, contraction = keys (contiguous in VpT rows)
        Frag vf0, vf1, vf2, vf3;
        const unsigned short* v0p = vbase + (size_t)( 0 + mrow) * 2048 + kt;
        const unsigned short* v1p = vbase + (size_t)(16 + mrow) * 2048 + kt;
        const unsigned short* v2p = vbase + (size_t)(32 + mrow) * 2048 + kt;
        const unsigned short* v3p = vbase + (size_t)(48 + mrow) * 2048 + kt;
        ((uint4*)vf0.u)[0] = *(const uint4*)(v0p + khalf);
        ((uint4*)vf0.u)[1] = *(const uint4*)(v0p + khalf + 16);
        ((uint4*)vf1.u)[0] = *(const uint4*)(v1p + khalf);
        ((uint4*)vf1.u)[1] = *(const uint4*)(v1p + khalf + 16);
        ((uint4*)vf2.u)[0] = *(const uint4*)(v2p + khalf);
        ((uint4*)vf2.u)[1] = *(const uint4*)(v2p + khalf + 16);
        ((uint4*)vf3.u)[0] = *(const uint4*)(v3p + khalf);
        ((uint4*)vf3.u)[1] = *(const uint4*)(v3p + khalf + 16);

        o0 = wmma_bf16(pa.v, vf0.v, o0);
        o1 = wmma_bf16(pa.v, vf1.v, o1);
        o2 = wmma_bf16(pa.v, vf2.v, o2);
        o3 = wmma_bf16(pa.v, vf3.v, o3);
        __syncthreads();
    }

    // finalize: O /= l, store bf16 to attnOut[b,s, h*64 + d]
    const size_t obase = ((size_t)(b * 2048 + qt * 16)) * 2048 + h * 64;
    #pragma unroll
    for (int r = 0; r < 8; r++) {
        const float inv = 1.0f / l[r];
        const size_t rb = obase + (size_t)(rowOff + r) * 2048;
        attnOut[rb +      mrow] = f2bf(o0[r] * inv);
        attnOut[rb + 16 + mrow] = f2bf(o1[r] * inv);
        attnOut[rb + 32 + mrow] = f2bf(o2[r] * inv);
        attnOut[rb + 48 + mrow] = f2bf(o3[r] * inv);
    }
}

// ---------------------------------------------------------------------------
extern "C" void kernel_launch(void* const* d_in, const int* in_sizes, int n_in,
                              void* d_out, int out_size, void* d_ws, size_t ws_size,
                              hipStream_t stream) {
    (void)in_sizes; (void)n_in; (void)out_size; (void)ws_size;

    const float* q  = (const float*)d_in[0];
    const float* k  = (const float*)d_in[1];
    const float* v  = (const float*)d_in[2];
    const float* Wq = (const float*)d_in[3];
    const float* bq = (const float*)d_in[4];
    const float* Wk = (const float*)d_in[5];
    const float* bk = (const float*)d_in[6];
    const float* Wv = (const float*)d_in[7];
    const float* bv = (const float*)d_in[8];
    const float* Wo = (const float*)d_in[9];
    const float* bo = (const float*)d_in[10];
    float* out = (float*)d_out;

    const int M  = 4096;          // B*S
    const int DM = 2048;
    const int KVD = 512;

    // workspace layout (ushort elements); total ~108 MB
    unsigned short* ws = (unsigned short*)d_ws;
    unsigned short* qb      = ws;                       // 4096*2048
    unsigned short* kb      = qb  + (size_t)M * DM;     // 4096*2048
    unsigned short* vb      = kb  + (size_t)M * DM;     // 4096*2048
    unsigned short* WqT     = vb  + (size_t)M * DM;     // 2048*2048
    unsigned short* WkT     = WqT + (size_t)DM * DM;    //  512*2048
    unsigned short* WvT     = WkT + (size_t)KVD * DM;   //  512*2048
    unsigned short* WoT     = WvT + (size_t)KVD * DM;   // 2048*2048
    unsigned short* Qp      = WoT + (size_t)DM * DM;    // 4096*2048
    unsigned short* Kp      = Qp  + (size_t)M * DM;     // 4096*512
    unsigned short* VpT     = Kp  + (size_t)M * KVD;    // 2*8*64*2048
    unsigned short* attnOut = VpT + (size_t)M * KVD;    // 4096*2048

    // 1) activations -> bf16
    {
        const int n = M * DM;
        const int g = (n + 255) / 256;
        cvt_bf16_kernel<<<g, 256, 0, stream>>>(q, qb, n);
        cvt_bf16_kernel<<<g, 256, 0, stream>>>(k, kb, n);
        cvt_bf16_kernel<<<g, 256, 0, stream>>>(v, vb, n);
    }
    // 2) weights -> transposed bf16
    transpose_bf16_kernel<<<(DM * DM + 255) / 256, 256, 0, stream>>>(Wq, WqT, DM, DM);
    transpose_bf16_kernel<<<(DM * KVD + 255) / 256, 256, 0, stream>>>(Wk, WkT, DM, KVD);
    transpose_bf16_kernel<<<(DM * KVD + 255) / 256, 256, 0, stream>>>(Wv, WvT, DM, KVD);
    transpose_bf16_kernel<<<(DM * DM + 255) / 256, 256, 0, stream>>>(Wo, WoT, DM, DM);

    // 3) projections (WMMA, async-LDS staged)
    gemm_bf16_kernel<0><<<dim3(M / 128, DM / 64), 256, 0, stream>>>(qb, WqT, bq, Qp,  M, DM,  DM);
    gemm_bf16_kernel<0><<<dim3(M / 128, KVD / 64), 256, 0, stream>>>(kb, WkT, bk, Kp,  M, KVD, DM);
    gemm_bf16_kernel<2><<<dim3(M / 128, KVD / 64), 256, 0, stream>>>(vb, WvT, bv, VpT, M, KVD, DM);

    // 4) flash attention: 2 batches * 32 heads * 128 query tiles
    gqa_attn_kernel<<<2 * 32 * 128, 32, 0, stream>>>(Qp, Kp, VpT, attnOut);

    // 5) output projection (WMMA, f32 out)
    gemm_bf16_kernel<1><<<dim3(M / 128, DM / 64), 256, 0, stream>>>(attnOut, WoT, bo, out, M, DM, DM);
}
